// QKV_Read_197568495701
// MI455X (gfx1250) — matrix operand order repaired and checked by reading, not compiled, not run on hardware
//
#include <hip/hip_runtime.h>
#include <hip/hip_bf16.h>

// ---------------------------------------------------------------------------
// CDNA5 (gfx1250) cross-attention pipeline, bf16 WMMA + async-to-LDS staging.
//   DIM=512, H=8, DH=64, LQ=16, LKV=128, B=1024
// ---------------------------------------------------------------------------

typedef __attribute__((ext_vector_type(16))) __bf16 v16bf;
typedef __attribute__((ext_vector_type(8)))  float  v8f;
typedef __attribute__((ext_vector_type(4)))  int    v4i;

union Frag16 { uint4 q[2]; v16bf v; };

// gfx1250 async global->LDS path (guarded; falls back to sync staging)
#if __has_builtin(__builtin_amdgcn_global_load_async_to_lds_b128) && \
    __has_builtin(__builtin_amdgcn_s_wait_asynccnt)
#define USE_ASYNC_LDS 1
#else
#define USE_ASYNC_LDS 0
#endif

__device__ __forceinline__ unsigned short f2bf(float f) {
  unsigned u = __float_as_uint(f);
  u += 0x7FFFu + ((u >> 16) & 1u);          // round-to-nearest-even
  return (unsigned short)(u >> 16);
}
__device__ __forceinline__ unsigned pack2(float a, float b) {
  return (unsigned)f2bf(a) | ((unsigned)f2bf(b) << 16);
}
__device__ __forceinline__ v8f v8f_zero() {
  v8f z;
#pragma unroll
  for (int i = 0; i < 8; ++i) z[i] = 0.0f;
  return z;
}
__device__ __forceinline__ v8f wmma_bf16(v16bf a, v16bf b, v8f c) {
  return __builtin_amdgcn_wmma_f32_16x16x32_bf16(
      /*neg_a=*/false, a, /*neg_b=*/false, b,
      /*c_mod=*/(short)0, c, /*reuse_a=*/false, /*reuse_b=*/false);
}

// ---------------------------------------------------------------------------
// Kernel 1: convert 4 weight matrices f32 [in][out] -> bf16 transposed [out][in]
// ---------------------------------------------------------------------------
__global__ __launch_bounds__(256) void prep_w(
    const float* __restrict__ Wq, const float* __restrict__ Wk,
    const float* __restrict__ Wv, const float* __restrict__ Wo,
    unsigned short* __restrict__ WT) {
  int idx = blockIdx.x * 256 + threadIdx.x;   // 0..262143
  int w = blockIdx.y;                         // which matrix
  const float* src = (w == 0) ? Wq : (w == 1) ? Wk : (w == 2) ? Wv : Wo;
  int i = idx >> 9;    // input dim (row of W)
  int o = idx & 511;   // output dim (col of W)
  WT[(long)w * 262144 + (long)o * 512 + i] = f2bf(src[idx]);
}

// ---------------------------------------------------------------------------
// Kernel 2: LayerNorm(x) -> bf16.  One wave per row of 512.
// ---------------------------------------------------------------------------
__global__ __launch_bounds__(256) void ln_kernel(
    const float* __restrict__ x, const float* __restrict__ gamma,
    const float* __restrict__ beta, unsigned short* __restrict__ xn) {
  int row  = blockIdx.x * 8 + (threadIdx.x >> 5);
  int lane = threadIdx.x & 31;
  const float* xr = x + (long)row * 512;
  float v[16];
  float s = 0.f, ss = 0.f;
#pragma unroll
  for (int j = 0; j < 16; ++j) {
    float t = xr[lane + j * 32];
    v[j] = t; s += t; ss += t * t;
  }
#pragma unroll
  for (int m = 16; m >= 1; m >>= 1) {
    s  += __shfl_xor(s,  m, 32);
    ss += __shfl_xor(ss, m, 32);
  }
  float mu  = s * (1.0f / 512.0f);
  float var = ss * (1.0f / 512.0f) - mu * mu;
  float rs  = rsqrtf(var + 1e-5f);
  unsigned short* xo = xn + (long)row * 512;
#pragma unroll
  for (int j = 0; j < 16; ++j) {
    int e = lane + j * 32;
    xo[e] = f2bf((v[j] - mu) * rs * gamma[e] + beta[e]);
  }
}

// ---------------------------------------------------------------------------
// Kernel 3: GEMM  C[M x 512] = A[M x 512] @ W[512 x 512] + bias (+residual)
//   WT is bf16 [out][in] so B-fragments are two 16B loads along a WT row.
//   A tile staged into double-buffered LDS, via GLOBAL_LOAD_ASYNC_TO_LDS_B128
//   (ASYNCcnt-pipelined) when available; raw f32 staged for the latents GEMMs
//   with conversion fused into A-fragment build (VALU co-executes with XDL).
//   Tiling: block 128x128, BK=32, 8 waves, each wave 2(M) x 4(N) WMMA tiles.
// ---------------------------------------------------------------------------
template <bool A_F32, bool OUT_RES>
__global__ __launch_bounds__(256) void gemm512(
    const void* __restrict__ Av, const unsigned short* __restrict__ WT,
    const float* __restrict__ bias, const float* __restrict__ resid,
    void* __restrict__ Cv) {
  constexpr int FSTR = 36;  // f32 LDS row stride (elems): 144B rows, 16B-aligned
  constexpr int HSTR = 48;  // bf16 LDS row stride (elems): 96B rows, 16B-aligned
  constexpr int TILEBYTES = A_F32 ? 128 * FSTR * 4 : 128 * HSTR * 2;
  __shared__ __align__(16) char Asmem[2 * TILEBYTES];

  const int lane = threadIdx.x & 31;
  const int wave = threadIdx.x >> 5;
  const int wm = wave & 3;    // row group: wm*32
  const int wn = wave >> 2;   // col group: wn*64
  const long rowBase = (long)blockIdx.x * 128;
  const int  colBase = blockIdx.y * 128;
  const int fr = lane & 15;   // fragment row/col within tile
  const int hi = lane >> 4;   // lane half
  const int srow = threadIdx.x >> 1;          // staging row 0..127
  const int scol = (threadIdx.x & 1) * 16;    // staging k-offset: 0 or 16 elems

  v8f acc[2][4];
#pragma unroll
  for (int mt = 0; mt < 2; ++mt)
#pragma unroll
    for (int nt = 0; nt < 4; ++nt) acc[mt][nt] = v8f_zero();

  // Stage one BK=32 tile slice for this thread into buffer `bufsel`.
  auto issue = [&](int kk, int bufsel) {
    char* base = Asmem + bufsel * TILEBYTES;
    if (A_F32) {
      const float* Af = (const float*)Av + (rowBase + srow) * 512 + kk + scol;
      char* dstb = base + srow * (FSTR * 4) + scol * 4;
#if USE_ASYNC_LDS
      auto g = (__attribute__((address_space(1))) v4i*)Af;
      auto l = (__attribute__((address_space(3))) v4i*)dstb;
      __builtin_amdgcn_global_load_async_to_lds_b128(g, l, 0, 0);
      __builtin_amdgcn_global_load_async_to_lds_b128(g, l, 16, 0);
      __builtin_amdgcn_global_load_async_to_lds_b128(g, l, 32, 0);
      __builtin_amdgcn_global_load_async_to_lds_b128(g, l, 48, 0);
#else
      const float4* s = (const float4*)Af;
      float4* dst = (float4*)dstb;
      dst[0] = s[0]; dst[1] = s[1]; dst[2] = s[2]; dst[3] = s[3];
      __builtin_prefetch(Af + 32, 0, 1);
#endif
    } else {
      const unsigned short* Ab =
          (const unsigned short*)Av + (rowBase + srow) * 512 + kk + scol;
      char* dstb = base + srow * (HSTR * 2) + scol * 2;
#if USE_ASYNC_LDS
      auto g = (__attribute__((address_space(1))) v4i*)Ab;
      auto l = (__attribute__((address_space(3))) v4i*)dstb;
      __builtin_amdgcn_global_load_async_to_lds_b128(g, l, 0, 0);
      __builtin_amdgcn_global_load_async_to_lds_b128(g, l, 16, 0);
#else
      const uint4* s = (const uint4*)Ab;
      uint4* dst = (uint4*)dstb;
      dst[0] = s[0]; dst[1] = s[1];
      __builtin_prefetch(Ab + 32, 0, 1);
#endif
    }
  };

  issue(0, 0);

  for (int k0 = 0; k0 < 512; k0 += 32) {
    const int buf = (k0 >> 5) & 1;
    const bool more = (k0 + 32) < 512;
    if (more) issue(k0 + 32, buf ^ 1);
#if USE_ASYNC_LDS
    // ASYNCcnt completes in order: waiting <= (loads just issued) guarantees
    // the previous tile's loads have landed in LDS.
    if (more) __builtin_amdgcn_s_wait_asynccnt(A_F32 ? 4 : 2);
    else      __builtin_amdgcn_s_wait_asynccnt(0);
#endif
    __syncthreads();

    const char* base = Asmem + buf * TILEBYTES;

    // ---- A fragments (16-bit A layout: per-lane kbase = hi*8) ----
    Frag16 af[2];
#pragma unroll
    for (int mt = 0; mt < 2; ++mt) {
      int r = wm * 32 + mt * 16 + fr;
      if (A_F32) {
        const float* Afl = (const float*)base;
        const float4* p0 = (const float4*)&Afl[r * FSTR + hi * 8];
        const float4* p1 = (const float4*)&Afl[r * FSTR + hi * 8 + 16];
        float4 a0 = p0[0], a1 = p0[1];
        float4 a2 = p1[0], a3 = p1[1];
        uint4 q0 = {pack2(a0.x, a0.y), pack2(a0.z, a0.w),
                    pack2(a1.x, a1.y), pack2(a1.z, a1.w)};
        uint4 q1 = {pack2(a2.x, a2.y), pack2(a2.z, a2.w),
                    pack2(a3.x, a3.y), pack2(a3.z, a3.w)};
        af[mt].q[0] = q0;
        af[mt].q[1] = q1;
      } else {
        const unsigned short* Ah = (const unsigned short*)base;
        af[mt].q[0] = *(const uint4*)&Ah[r * HSTR + hi * 8];
        af[mt].q[1] = *(const uint4*)&Ah[r * HSTR + hi * 8 + 16];
      }
    }
    // ---- B fragments from global WT (16B chunks: kbase = hi*8, +16) ----
#pragma unroll
    for (int nt = 0; nt < 4; ++nt) {
      int nc = colBase + wn * 64 + nt * 16 + fr;
      const uint4* bp = (const uint4*)&WT[(long)nc * 512 + k0 + hi * 8];
      Frag16 bf_;
      bf_.q[0] = bp[0];
      bf_.q[1] = bp[2];
#pragma unroll
      for (int mt = 0; mt < 2; ++mt)
        acc[mt][nt] = wmma_bf16(af[mt].v, bf_.v, acc[mt][nt]);
    }
    __syncthreads();
  }

  // ---- epilogue: C-layout (lane<16: N=lane, M=i ; lane>=16: M=i+8) ----
#pragma unroll
  for (int mt = 0; mt < 2; ++mt) {
#pragma unroll
    for (int nt = 0; nt < 4; ++nt) {
      int col = colBase + wn * 64 + nt * 16 + fr;
      float bvv = bias[col];
      long r0 = rowBase + wm * 32 + mt * 16 + hi * 8;
      if (OUT_RES) {
        float* C = (float*)Cv;
#pragma unroll
        for (int i = 0; i < 8; ++i) {
          long r = r0 + i;
          C[r * 512 + col] = acc[mt][nt][i] + bvv + resid[r * 512 + col];
        }
      } else {
        unsigned short* C = (unsigned short*)Cv;
#pragma unroll
        for (int i = 0; i < 8; ++i) {
          long r = r0 + i;
          C[r * 512 + col] = f2bf(acc[mt][nt][i] + bvv);
        }
      }
    }
  }
}

// ---------------------------------------------------------------------------
// Kernel 4: attention core. One wave (32 threads) per (b,h) pair.
//   scores = Q_h(16x64) @ K_h^T * scale + rel_bias -> softmax ->
//   ctx_h  = P(16x128) @ V_h(128x64), all via v_wmma_f32_16x16x32_bf16.
// ---------------------------------------------------------------------------
#define VSTR 136   // LDS stride (elems); 272B rows keep uint4 alignment

__global__ __launch_bounds__(32) void attn_kernel(
    const unsigned short* __restrict__ Q, const unsigned short* __restrict__ K,
    const unsigned short* __restrict__ V, const float* __restrict__ rel_bias,
    unsigned short* __restrict__ ctx) {
  __shared__ __align__(16) unsigned short Vt[64 * VSTR];  // V_h transposed [dh][kv]
  __shared__ __align__(16) unsigned short Pl[16 * VSTR];  // probs bf16 [q][kv]

  const int lane = threadIdx.x;
  const int bh = blockIdx.x;
  const int b = bh >> 3, h = bh & 7;
  const long kvBase = (long)b * 128 * 512 + h * 64;  // base of K_h/V_h rows

  // ---- stage V_h transposed into LDS: Vt[dh][kv] ----
  for (int it = 0; it < 32; ++it) {
    int idx = it * 32 + lane;        // 0..1023
    int r   = idx >> 3;              // kv row 0..127
    int c8  = (idx & 7) * 8;         // dh chunk base
    uint4 d = *(const uint4*)&V[kvBase + (long)r * 512 + c8];
    unsigned u[4] = {d.x, d.y, d.z, d.w};
#pragma unroll
    for (int j = 0; j < 4; ++j) {
      Vt[(c8 + 2 * j)     * VSTR + r] = (unsigned short)(u[j] & 0xFFFFu);
      Vt[(c8 + 2 * j + 1) * VSTR + r] = (unsigned short)(u[j] >> 16);
    }
  }
  __syncthreads();

  const int fr = lane & 15, hi = lane >> 4;

  // ---- Q fragments (A layout), direct from global ----
  Frag16 qf[2];
  {
    const unsigned short* qp = Q + (long)(b * 16 + fr) * 512 + h * 64 + hi * 8;
    qf[0].q[0] = *(const uint4*)qp;
    qf[0].q[1] = *(const uint4*)(qp + 16);
    qf[1].q[0] = *(const uint4*)(qp + 32);
    qf[1].q[1] = *(const uint4*)(qp + 48);
  }

  // ---- scores: S[16 x 128] = Q @ K^T ----
  v8f sacc[8];
#pragma unroll
  for (int nt = 0; nt < 8; ++nt) sacc[nt] = v8f_zero();
#pragma unroll
  for (int nt = 0; nt < 8; ++nt) {
#pragma unroll
    for (int kk = 0; kk < 2; ++kk) {
      // B = K_h^T : B-frag lane column n -> row (nt*16+n) of K_h, k = hi*8 (+16)
      const unsigned short* kp =
          K + kvBase + (long)(nt * 16 + fr) * 512 + kk * 32 + hi * 8;
      Frag16 bf_;
      bf_.q[0] = ((const uint4*)kp)[0];
      bf_.q[1] = ((const uint4*)kp)[2];
      sacc[nt] = wmma_bf16(qf[kk].v, bf_.v, sacc[nt]);
    }
  }

  // ---- scale + relative position bias ----
  const float scale = 0.125f;  // 1/sqrt(64)
  const float* rb = rel_bias + (long)h * 16 * 128;
#pragma unroll
  for (int nt = 0; nt < 8; ++nt)
#pragma unroll
    for (int i = 0; i < 8; ++i) {
      int m = i + hi * 8, c = nt * 16 + fr;
      sacc[nt][i] = sacc[nt][i] * scale + rb[m * 128 + c];
    }

  // ---- softmax over kv (row m lives in one 16-lane group across 8 regs) ----
#pragma unroll
  for (int i = 0; i < 8; ++i) {
    float mx = sacc[0][i];
#pragma unroll
    for (int nt = 1; nt < 8; ++nt) mx = fmaxf(mx, sacc[nt][i]);
#pragma unroll
    for (int msk = 1; msk < 16; msk <<= 1) mx = fmaxf(mx, __shfl_xor(mx, msk, 32));
    float sum = 0.f;
#pragma unroll
    for (int nt = 0; nt < 8; ++nt) {
      float e = __expf(sacc[nt][i] - mx);
      sacc[nt][i] = e;
      sum += e;
    }
#pragma unroll
    for (int msk = 1; msk < 16; msk <<= 1) sum += __shfl_xor(sum, msk, 32);
    float inv = 1.0f / sum;
#pragma unroll
    for (int nt = 0; nt < 8; ++nt) sacc[nt][i] *= inv;
  }

  // ---- P -> LDS as bf16 (row-major so A-frag loads are contiguous) ----
#pragma unroll
  for (int nt = 0; nt < 8; ++nt)
#pragma unroll
    for (int i = 0; i < 8; ++i) {
      int m = i + hi * 8, c = nt * 16 + fr;
      Pl[m * VSTR + c] = f2bf(sacc[nt][i]);
    }
  __syncthreads();

  // ---- ctx_h = P @ V_h ----
  v8f oacc[4];
#pragma unroll
  for (int nt = 0; nt < 4; ++nt) oacc[nt] = v8f_zero();
#pragma unroll
  for (int kk = 0; kk < 4; ++kk) {
    Frag16 af;
    const unsigned short* pp = &Pl[fr * VSTR + kk * 32 + hi * 8];
    af.q[0] = *(const uint4*)pp;
    af.q[1] = *(const uint4*)(pp + 16);
#pragma unroll
    for (int nt = 0; nt < 4; ++nt) {
      const unsigned short* vp = &Vt[(nt * 16 + fr) * VSTR + kk * 32 + hi * 8];
      Frag16 bf_;
      bf_.q[0] = ((const uint4*)vp)[0];
      bf_.q[1] = ((const uint4*)vp)[2];
      oacc[nt] = wmma_bf16(af.v, bf_.v, oacc[nt]);
    }
  }

  // ---- store context bf16: ctx[b*16+m][h*64 + dh] ----
#pragma unroll
  for (int nt = 0; nt < 4; ++nt)
#pragma unroll
    for (int i = 0; i < 8; ++i) {
      int m = i + hi * 8;
      ctx[(long)(b * 16 + m) * 512 + h * 64 + nt * 16 + fr] = f2bf(oacc[nt][i]);
    }
}

// ---------------------------------------------------------------------------
// Host-side launch
// ---------------------------------------------------------------------------
extern "C" void kernel_launch(void* const* d_in, const int* in_sizes, int n_in,
                              void* d_out, int out_size, void* d_ws,
                              size_t ws_size, hipStream_t stream) {
  (void)in_sizes; (void)n_in; (void)out_size; (void)ws_size;
  const float* x        = (const float*)d_in[0];
  const float* latents  = (const float*)d_in[1];
  const float* Wq       = (const float*)d_in[2];
  const float* bq       = (const float*)d_in[3];
  const float* Wk       = (const float*)d_in[4];
  const float* bk       = (const float*)d_in[5];
  const float* Wv       = (const float*)d_in[6];
  const float* bv       = (const float*)d_in[7];
  const float* Wo       = (const float*)d_in[8];
  const float* bo       = (const float*)d_in[9];
  const float* gamma    = (const float*)d_in[10];
  const float* beta     = (const float*)d_in[11];
  const float* rel_bias = (const float*)d_in[12];

  // Workspace carve-up (bf16 intermediates), ~321 MB total.
  char* ws = (char*)d_ws;
  unsigned short* WT  = (unsigned short*)ws; ws += 4l * 512 * 512 * 2;      // 2 MB
  unsigned short* xn  = (unsigned short*)ws; ws += 16384l * 512 * 2;        // 16 MB
  unsigned short* qb  = (unsigned short*)ws; ws += 16384l * 512 * 2;        // 16 MB
  unsigned short* ctx = (unsigned short*)ws; ws += 16384l * 512 * 2;        // 16 MB
  unsigned short* kb  = (unsigned short*)ws; ws += 131072l * 512 * 2;       // 134 MB
  unsigned short* vb  = (unsigned short*)ws; ws += 131072l * 512 * 2;       // 134 MB

  prep_w<<<dim3(1024, 4), 256, 0, stream>>>(Wq, Wk, Wv, Wo, WT);
  ln_kernel<<<2048, 256, 0, stream>>>(x, gamma, beta, xn);

  // Q = LN(x) @ Wq + bq            (M = 16384)
  gemm512<false, false><<<dim3(128, 4), 256, 0, stream>>>(
      (const void*)xn, WT + 0l * 262144, bq, nullptr, (void*)qb);
  // K = latents @ Wk + bk          (M = 131072, f32 A async-staged, cvt at frag)
  gemm512<true, false><<<dim3(1024, 4), 256, 0, stream>>>(
      (const void*)latents, WT + 1l * 262144, bk, nullptr, (void*)kb);
  // V = latents @ Wv + bv
  gemm512<true, false><<<dim3(1024, 4), 256, 0, stream>>>(
      (const void*)latents, WT + 2l * 262144, bv, nullptr, (void*)vb);

  // attention core: one wave per (b, h)
  attn_kernel<<<8192, 32, 0, stream>>>(qb, kb, vb, rel_bias, ctx);

  // out = ctx @ Wo + bo + x        (f32 output with residual)
  gemm512<false, true><<<dim3(128, 4), 256, 0, stream>>>(
      (const void*)ctx, WT + 3l * 262144, bo, x, d_out);
}
